// Attention_52991306498338
// MI455X (gfx1250) — compile-verified, matrix operand order checked by reading, not run on hardware
//
#include <hip/hip_runtime.h>
#include <hip/hip_bf16.h>
#include <stdint.h>

#define B_ 4
#define S_ 2048
#define E_ 1024

typedef __attribute__((ext_vector_type(16))) __bf16 v16bf;
typedef __attribute__((ext_vector_type(8)))  float  v8f;

union Frag {            // 16 bf16 = 32 bytes = 8 VGPRs (A or B fragment)
  v16bf v;
  uint4 q[2];
};
union AccU {            // 16x16 f32 C/D tile fragment: 8 VGPRs
  v8f   v;
  float f[8];
};

__device__ __forceinline__ unsigned short f2bf(float x) {
  unsigned u = __float_as_uint(x);
  u += 0x7FFFu + ((u >> 16) & 1u);      // round-to-nearest-even
  return (unsigned short)(u >> 16);
}

__device__ __forceinline__ v8f wmma_bf16(const Frag& a, const Frag& b, v8f c) {
  return __builtin_amdgcn_wmma_f32_16x16x32_bf16(
      false, a.v, false, b.v, (short)0, c, false, false);
}

// --------------------------- TDM support -----------------------------------
#if defined(__has_builtin)
#  if __has_builtin(__builtin_amdgcn_tensor_load_to_lds) && \
      __has_builtin(__builtin_amdgcn_s_wait_tensorcnt)
#    define USE_TDM 1
#  endif
#endif
#ifndef USE_TDM
#  define USE_TDM 0
#endif

#if USE_TDM
typedef __attribute__((ext_vector_type(4))) unsigned int u32x4;
typedef __attribute__((ext_vector_type(8))) int          i32x8;
typedef __attribute__((ext_vector_type(4))) int          i32x4;

// 2D bf16 tile load: tile_dim0 = 32 (contiguous k), tile_dim1 = 128 rows.
// D# packing per CDNA5 ISA (groups 0 and 1).
// clang-23 toolchain: 6-arg builtin (g0, g1, g2, g3, g4, cpol).
__device__ __forceinline__ void tdm_load_tile_bf16(
    unsigned lds_addr, const void* gptr,
    unsigned tensor_d0, unsigned tensor_d1, unsigned stride0)
{
  unsigned long long ga = (unsigned long long)(size_t)gptr;
  u32x4 g0;
  g0.x = 0x1u;                                   // count=1, user descriptor
  g0.y = lds_addr;                               // lds_addr [63:32]
  g0.z = (unsigned)(ga & 0xFFFFFFFFu);           // global_addr [95:64]
  g0.w = (unsigned)((ga >> 32) & 0x01FFFFFFu)    // global_addr [120:96]
       | (2u << 30);                             // type=2 ("image")
  i32x8 g1;
  g1[0] = 0x00010000;                            // data_size=1 -> 2 bytes
  g1[1] = (int)((tensor_d0 & 0xFFFFu) << 16);    // tensor_dim0[15:0] @ bit48
  g1[2] = (int)(((tensor_d0 >> 16) & 0xFFFFu) |
                ((tensor_d1 & 0xFFFFu) << 16));  // dim0[31:16] | dim1[15:0]
  g1[3] = (int)(((tensor_d1 >> 16) & 0xFFFFu) |
                (32u << 16));                    // dim1[31:16] | tile_dim0=32
  g1[4] = (int)128;                              // tile_dim1=128, tile_dim2=0
  g1[5] = (int)stride0;                          // tensor_dim0_stride[31:0]
  g1[6] = 0;                                     // stride0[47:32], d1stride lo
  g1[7] = 0;
  i32x4 z4 = {0, 0, 0, 0};                       // groups 2/3: unused (2D)
  i32x8 z8 = {0, 0, 0, 0, 0, 0, 0, 0};
  __builtin_amdgcn_tensor_load_to_lds(g0, g1, z4, z4, z8, 0);
}
#endif

// ---------------------------------------------------------------------------
// Kernel 0: f32 -> bf16 bulk convert (x and the three weight matrices), so
// the GEMM streams bf16 (half the L2 bytes) and the inner loop has no cvt.
// ---------------------------------------------------------------------------
__global__ __launch_bounds__(256) void cvt_bf16_kernel(
    const float* __restrict__ src, unsigned short* __restrict__ dst)
{
  size_t i = ((size_t)blockIdx.x * 256 + threadIdx.x) * 8;
  float4 a = *(const float4*)(src + i);
  float4 b = *(const float4*)(src + i + 4);
  uint4 pk;
  pk.x = (unsigned)f2bf(a.x) | ((unsigned)f2bf(a.y) << 16);
  pk.y = (unsigned)f2bf(a.z) | ((unsigned)f2bf(a.w) << 16);
  pk.z = (unsigned)f2bf(b.x) | ((unsigned)f2bf(b.y) << 16);
  pk.w = (unsigned)f2bf(b.z) | ((unsigned)f2bf(b.w) << 16);
  *(uint4*)(dst + i) = pk;
}

// ---------------------------------------------------------------------------
// Kernel 1: fused Q/K/V projection, bf16 in / bf16 out, f32 accumulate.
// O = x @ W^T + b.  blockIdx.z selects {Q,K,V}.  Q,K row-major; V transposed.
// 128x128 tile, k-step 32, 8 waves x (2x4) WMMA C tiles.
// Tile staging: TDM tensor_load_to_lds, double buffered (fallback: ds copy).
// ---------------------------------------------------------------------------
__global__ __launch_bounds__(256) void qkv_proj_kernel(
    const unsigned short* __restrict__ xb,   // bf16 [B*S][E]
    const unsigned short* __restrict__ Wb,   // bf16 [3][E][E]
    const float* __restrict__ bq, const float* __restrict__ bk,
    const float* __restrict__ bv,
    unsigned short* __restrict__ Qws,
    unsigned short* __restrict__ Kws,
    unsigned short* __restrict__ Vt)
{
  __shared__ unsigned short sX[2][128][32];   // 2 x 8KB
  __shared__ unsigned short sW[2][128][32];   // 2 x 8KB

  const int z = blockIdx.z;
  const float* bias = (z == 0) ? bq : (z == 1) ? bk : bv;
  const unsigned short* Wz = Wb + (size_t)z * E_ * E_;

  const int rows0 = blockIdx.x * 128;
  const int cols0 = blockIdx.y * 128;
  const int tid  = threadIdx.x;
  const int wid  = tid >> 5;
  const int lane = tid & 31;
  const int hi   = lane >> 4;
  const int l16  = lane & 15;
  const int wm = (wid & 3) * 32;
  const int wn = (wid >> 2) * 64;

  AccU acc[2][4];
  #pragma unroll
  for (int i = 0; i < 2; ++i)
    #pragma unroll
    for (int j = 0; j < 4; ++j)
      #pragma unroll
      for (int r = 0; r < 8; ++r) acc[i][j].f[r] = 0.0f;

#if USE_TDM
  if (tid < 32) {   // wave 0 issues DMA; TDM ignores EXEC, one issue per tile
    tdm_load_tile_bf16((unsigned)(size_t)&sX[0][0][0],
                       xb + (size_t)rows0 * E_, E_, B_ * S_, E_);
    tdm_load_tile_bf16((unsigned)(size_t)&sW[0][0][0],
                       Wz + (size_t)cols0 * E_, E_, E_, E_);
  }
#endif

  for (int kk = 0; kk < E_; kk += 32) {
    const int buf = (kk >> 5) & 1;

#if USE_TDM
    if (tid < 32) {
      if (kk + 32 < E_) {   // prefetch next tile into the other buffer
        tdm_load_tile_bf16((unsigned)(size_t)&sX[buf ^ 1][0][0],
                           xb + (size_t)rows0 * E_ + kk + 32, E_, B_ * S_, E_);
        tdm_load_tile_bf16((unsigned)(size_t)&sW[buf ^ 1][0][0],
                           Wz + (size_t)cols0 * E_ + kk + 32, E_, E_, E_);
        __builtin_amdgcn_s_wait_tensorcnt(2);   // current pair landed
      } else {
        __builtin_amdgcn_s_wait_tensorcnt(0);
      }
    }
    __syncthreads();
#else
    // synchronous fallback staging: 256 threads x 2 x b128 per tile
    #pragma unroll
    for (int it = 0; it < 2; ++it) {
      int j = tid * 2 + it;          // uint4 index, 512 per tile
      int r = j >> 2;
      int c = (j & 3) * 8;
      *(uint4*)&sX[buf][r][c] =
          *(const uint4*)(xb + (size_t)(rows0 + r) * E_ + kk + c);
      *(uint4*)&sW[buf][r][c] =
          *(const uint4*)(Wz + (size_t)(cols0 + r) * E_ + kk + c);
    }
    __syncthreads();
#endif

    Frag bfr[4];
    #pragma unroll
    for (int ni = 0; ni < 4; ++ni) {
      int wr = wn + ni * 16 + l16;
      bfr[ni].q[0] = *(const uint4*)&sW[buf][wr][hi * 8];
      bfr[ni].q[1] = *(const uint4*)&sW[buf][wr][16 + hi * 8];
    }
    #pragma unroll
    for (int mi = 0; mi < 2; ++mi) {
      Frag afr;
      int xr = wm + mi * 16 + l16;
      afr.q[0] = *(const uint4*)&sX[buf][xr][hi * 8];
      afr.q[1] = *(const uint4*)&sX[buf][xr][16 + hi * 8];
      #pragma unroll
      for (int ni = 0; ni < 4; ++ni)
        acc[mi][ni].v = wmma_bf16(afr, bfr[ni], acc[mi][ni].v);
    }
    __syncthreads();
  }

  // epilogue: bias add, bf16 convert, store
  if (z < 2) {
    unsigned short* out = (z == 0) ? Qws : Kws;
    #pragma unroll
    for (int mi = 0; mi < 2; ++mi) {
      #pragma unroll
      for (int ni = 0; ni < 4; ++ni) {
        int col = cols0 + wn + ni * 16 + l16;
        float bb = bias[col];
        #pragma unroll
        for (int r = 0; r < 8; ++r) {
          int row = rows0 + wm + mi * 16 + hi * 8 + r;
          out[(size_t)row * E_ + col] = f2bf(acc[mi][ni].f[r] + bb);
        }
      }
    }
  } else {
    // V stored transposed: Vt[b][col][s]; 8 consecutive s -> one b128 store
    #pragma unroll
    for (int mi = 0; mi < 2; ++mi) {
      int rowbase = rows0 + wm + mi * 16 + hi * 8;
      int bb_ = rowbase / S_;
      int s   = rowbase % S_;
      #pragma unroll
      for (int ni = 0; ni < 4; ++ni) {
        int col = cols0 + wn + ni * 16 + l16;
        float bv_ = bias[col];
        unsigned short h[8];
        #pragma unroll
        for (int r = 0; r < 8; ++r) h[r] = f2bf(acc[mi][ni].f[r] + bv_);
        uint4 pk;
        pk.x = (unsigned)h[0] | ((unsigned)h[1] << 16);
        pk.y = (unsigned)h[2] | ((unsigned)h[3] << 16);
        pk.z = (unsigned)h[4] | ((unsigned)h[5] << 16);
        pk.w = (unsigned)h[6] | ((unsigned)h[7] << 16);
        *(uint4*)&Vt[((size_t)bb_ * E_ + col) * S_ + s] = pk;
      }
    }
  }
}

// ---------------------------------------------------------------------------
// Kernel 2: flash attention. 16-query block per workgroup (8 waves); wave w
// owns e-slice [128w,128w+128) for the score reduction and output columns.
// ---------------------------------------------------------------------------
__global__ __launch_bounds__(256) void flash_attn_kernel(
    const unsigned short* __restrict__ Qws,
    const unsigned short* __restrict__ Kws,
    const unsigned short* __restrict__ Vt,
    float* __restrict__ out)
{
  __shared__ float part[8][16][32];
  __shared__ float sc[16][32];
  __shared__ unsigned short sp[16][32];
  __shared__ float sm[16], sl[16], scor[16];

  const int b  = blockIdx.y;
  const int q0 = blockIdx.x * 16;
  const int tid = threadIdx.x;
  const int wid = tid >> 5, lane = tid & 31, hi = lane >> 4, l16 = lane & 15;
  const int es0 = wid * 128;
  const float scale = 0.03125f;              // 1/sqrt(1024)

  if (tid < 16) { sm[tid] = -3.0e38f; sl[tid] = 0.0f; }

  Frag qf[4];
  {
    const unsigned short* qrow = Qws + (size_t)(b * S_ + q0 + l16) * E_;
    #pragma unroll
    for (int c = 0; c < 4; ++c) {
      qf[c].q[0] = *(const uint4*)(qrow + es0 + c * 32 + hi * 8);
      qf[c].q[1] = *(const uint4*)(qrow + es0 + c * 32 + 16 + hi * 8);
    }
  }

  AccU oacc[8];
  #pragma unroll
  for (int t = 0; t < 8; ++t)
    #pragma unroll
    for (int r = 0; r < 8; ++r) oacc[t].f[r] = 0.0f;

  __syncthreads();

  const unsigned short* kb0 = Kws + (size_t)(b * S_) * E_;

  for (int k0 = 0; k0 < S_; k0 += 32) {
    // prefetch next key/value tiles (L2-resident; warm the near caches)
    {
      int kp = (k0 + 32 < S_) ? (k0 + 32) : k0;
      __builtin_prefetch(kb0 + (size_t)(kp + l16) * E_ + es0, 0, 1);
      __builtin_prefetch(kb0 + (size_t)(kp + 16 + l16) * E_ + es0, 0, 1);
      __builtin_prefetch(Vt + ((size_t)b * E_ + es0 + lane * 4) * S_ + kp, 0, 1);
    }

    // ---- partial scores over this wave's e-slice
    AccU s0, s1;
    #pragma unroll
    for (int r = 0; r < 8; ++r) { s0.f[r] = 0.0f; s1.f[r] = 0.0f; }
    #pragma unroll
    for (int c = 0; c < 4; ++c) {
      Frag kb;
      const unsigned short* kr = kb0 + (size_t)(k0 + l16) * E_ + es0 + c * 32;
      kb.q[0] = *(const uint4*)(kr + hi * 8);
      kb.q[1] = *(const uint4*)(kr + 16 + hi * 8);
      s0.v = wmma_bf16(qf[c], kb, s0.v);
      const unsigned short* kr2 = kr + (size_t)16 * E_;
      kb.q[0] = *(const uint4*)(kr2 + hi * 8);
      kb.q[1] = *(const uint4*)(kr2 + 16 + hi * 8);
      s1.v = wmma_bf16(qf[c], kb, s1.v);
    }
    #pragma unroll
    for (int r = 0; r < 8; ++r) {
      part[wid][hi * 8 + r][l16]      = s0.f[r];
      part[wid][hi * 8 + r][16 + l16] = s1.f[r];
    }
    __syncthreads();

    // ---- cross-wave reduction
    {
      int idx = tid * 2;
      int row = idx >> 5, col = idx & 31;
      float a0 = 0.0f, a1 = 0.0f;
      #pragma unroll
      for (int w = 0; w < 8; ++w) { a0 += part[w][row][col]; a1 += part[w][row][col + 1]; }
      sc[row][col]     = a0 * scale;
      sc[row][col + 1] = a1 * scale;
    }
    __syncthreads();

    // ---- online softmax update (one thread per query row)
    if (tid < 16) {
      float mOld = sm[tid];
      float mx = mOld;
      #pragma unroll
      for (int j = 0; j < 32; ++j) mx = fmaxf(mx, sc[tid][j]);
      float corr = __expf(mOld - mx);
      float l = sl[tid] * corr;
      #pragma unroll
      for (int j = 0; j < 32; ++j) {
        float p = __expf(sc[tid][j] - mx);
        l += p;
        sp[tid][j] = f2bf(p);
      }
      sm[tid] = mx; sl[tid] = l; scor[tid] = corr;
    }
    __syncthreads();

    // ---- rescale accumulators
    float cr[8];
    #pragma unroll
    for (int r = 0; r < 8; ++r) cr[r] = scor[hi * 8 + r];
    #pragma unroll
    for (int t = 0; t < 8; ++t)
      #pragma unroll
      for (int r = 0; r < 8; ++r) oacc[t].f[r] *= cr[r];

    // ---- P @ V
    Frag pf;
    pf.q[0] = *(const uint4*)&sp[l16][hi * 8];
    pf.q[1] = *(const uint4*)&sp[l16][16 + hi * 8];
    #pragma unroll
    for (int ni = 0; ni < 8; ++ni) {
      int col = es0 + ni * 16 + l16;
      const unsigned short* vr = Vt + ((size_t)b * E_ + col) * S_ + k0;
      Frag vb;
      vb.q[0] = *(const uint4*)(vr + hi * 8);
      vb.q[1] = *(const uint4*)(vr + 16 + hi * 8);
      oacc[ni].v = wmma_bf16(pf, vb, oacc[ni].v);
    }
    __syncthreads();
  }

  // ---- finalize
  float invl[8];
  #pragma unroll
  for (int r = 0; r < 8; ++r) invl[r] = 1.0f / sl[hi * 8 + r];
  #pragma unroll
  for (int ni = 0; ni < 8; ++ni) {
    int col = es0 + ni * 16 + l16;
    #pragma unroll
    for (int r = 0; r < 8; ++r) {
      int q = q0 + hi * 8 + r;
      out[((size_t)(b * S_ + q)) * E_ + col] = oacc[ni].f[r] * invl[r];
    }
  }
}

// ---------------------------------------------------------------------------
extern "C" void kernel_launch(void* const* d_in, const int* in_sizes, int n_in,
                              void* d_out, int out_size, void* d_ws, size_t ws_size,
                              hipStream_t stream) {
  (void)in_sizes; (void)n_in; (void)out_size; (void)ws_size;
  const float* x  = (const float*)d_in[0];
  const float* Wq = (const float*)d_in[1];
  const float* bq = (const float*)d_in[2];
  const float* Wk = (const float*)d_in[3];
  const float* bk = (const float*)d_in[4];
  const float* Wv = (const float*)d_in[5];
  const float* bv = (const float*)d_in[6];
  float* out = (float*)d_out;

  // Workspace: Q | K | V^T (bf16, 16MB each) | x_bf16 (16MB) | W_bf16 (6MB)
  const size_t NME = (size_t)B_ * S_ * E_;   // 8M elements
  const size_t NWE = (size_t)E_ * E_;        // 1M elements
  unsigned short* Qws = (unsigned short*)d_ws;
  unsigned short* Kws = Qws + NME;
  unsigned short* Vt  = Kws + NME;
  unsigned short* xb  = Vt + NME;
  unsigned short* Wb  = xb + NME;            // [3][E][E]

  // 0) bulk f32 -> bf16 conversion
  cvt_bf16_kernel<<<dim3((unsigned)(NME / 2048)), 256, 0, stream>>>(x, xb);
  cvt_bf16_kernel<<<dim3((unsigned)(NWE / 2048)), 256, 0, stream>>>(Wq, Wb);
  cvt_bf16_kernel<<<dim3((unsigned)(NWE / 2048)), 256, 0, stream>>>(Wk, Wb + NWE);
  cvt_bf16_kernel<<<dim3((unsigned)(NWE / 2048)), 256, 0, stream>>>(Wv, Wb + 2 * NWE);

  // 1) Q/K/V projection (TDM double-buffered tiles)
  dim3 g1(B_ * S_ / 128, E_ / 128, 3);
  qkv_proj_kernel<<<g1, 256, 0, stream>>>(xb, Wb, bq, bk, bv, Qws, Kws, Vt);

  // 2) flash attention
  dim3 g2(S_ / 16, B_);
  flash_attn_kernel<<<g2, 256, 0, stream>>>(Qws, Kws, Vt, out);
}
